// MultiAtomAttention_48155173323470
// MI455X (gfx1250) — matrix operand order repaired and checked by reading, not compiled, not run on hardware
//
#include <hip/hip_runtime.h>
#include <hip/hip_bf16.h>

#define EMB 1024
#define HEADS 4
#define ATT 256
#define SCALE 0.0625f   // 256^-0.5
#define LN_EPS 1e-5f

typedef __bf16 bf16_t;
typedef __attribute__((ext_vector_type(16))) __bf16 v16bf;
typedef __attribute__((ext_vector_type(8)))  float  v8f;

union ABFrag { v16bf v; uint4 q[2]; };

static __device__ inline bf16_t f2bf(float f) {
    union { float f; unsigned u; } x; x.f = f;
    unsigned r = x.u + 0x7FFFu + ((x.u >> 16) & 1u);   // round-to-nearest-even
    unsigned short h = (unsigned short)(r >> 16);
    bf16_t b; __builtin_memcpy(&b, &h, 2);
    return b;
}

// ---------------- fp32 -> bf16 convert ----------------
__global__ void cvt_bf16_kernel(const float* __restrict__ src, bf16_t* __restrict__ dst, int n) {
    int i = blockIdx.x * blockDim.x + threadIdx.x;
    if (i < n) dst[i] = f2bf(src[i]);
}

// ---------------- 32x32 LDS-tiled transpose + convert: WT[c][r] = W[r][c] ----
__global__ void transpose_cvt_kernel(const float* __restrict__ W, bf16_t* __restrict__ WT,
                                     int rows, int cols) {
    __shared__ float tile[32][33];
    int c0 = blockIdx.x * 32, r0 = blockIdx.y * 32;
    int tx = threadIdx.x & 31, ty = threadIdx.x >> 5;         // 256 thr: ty 0..7
    for (int i = ty; i < 32; i += 8)
        tile[i][tx] = W[(size_t)(r0 + i) * cols + c0 + tx];
    __syncthreads();
    for (int i = ty; i < 32; i += 8)
        WT[(size_t)(c0 + i) * rows + r0 + tx] = f2bf(tile[tx][i]);
}

// ---------------- WMMA fragment loads -----------------
// A (16x32, row-major, lda elems): lane<16 -> row=lane, K {0..7,16..23}
//                                  lane>=16 -> row=lane-16, K {8..15,24..31}
static __device__ inline v16bf load_a_frag(const bf16_t* p, int lda, int lane) {
    int half = lane >> 4, r = lane & 15;
    const bf16_t* q = p + (size_t)r * lda + (half << 3);
    ABFrag u;
    u.q[0] = *(const uint4*)q;
    u.q[1] = *(const uint4*)(q + 16);
    return u.v;
}
// B (32x16) supplied as BT rows (N x K row-major, ldb elems):
// lane<16 -> col=lane, K {0..7,16..23}; lane>=16 -> col=lane-16, K {8..15,24..31}
static __device__ inline v16bf load_b_frag(const bf16_t* p, int ldb, int lane) {
    int half = lane >> 4, c = lane & 15;
    const bf16_t* q = p + (size_t)c * ldb + (half << 3);
    ABFrag u;
    u.q[0] = *(const uint4*)q;
    u.q[1] = *(const uint4*)(q + 16);
    return u.v;
}

// ---------------- bf16 WMMA GEMM: C[MxN] = A[MxK] * BT[NxK]^T -----------------
// MODE 0: fp32 out (+ optional bias[n])
// MODE 1: bf16 out, row-major
// MODE 2: bf16 out, transposed (CT[n*ldc+m])
// MODE 3: relu + bf16 out, row-major
// Block = 256 threads = 8 waves (2x4), wave tile 64x64 -> block tile 128x256.
template <int MODE>
__global__ __launch_bounds__(256)
void gemm_bf16_wmma(const bf16_t* __restrict__ A, int lda,
                    const bf16_t* __restrict__ BT, int ldb,
                    void* __restrict__ C, int ldc,
                    int K, const float* __restrict__ bias) {
    const int lane = threadIdx.x & 31;
    const int wave = threadIdx.x >> 5;
    const int wm = wave >> 2;            // 0..1
    const int wn = wave & 3;             // 0..3
    const int m0 = blockIdx.y * 128 + wm * 64;
    const int n0 = blockIdx.x * 256 + wn * 64;

    v8f acc[4][4];
    for (int i = 0; i < 4; ++i)
        for (int j = 0; j < 4; ++j) { v8f z = {}; acc[i][j] = z; }

    for (int k0 = 0; k0 < K; k0 += 32) {
        v16bf af[4], bf[4];
#pragma unroll
        for (int i = 0; i < 4; ++i)
            af[i] = load_a_frag(A + (size_t)(m0 + 16 * i) * lda + k0, lda, lane);
#pragma unroll
        for (int j = 0; j < 4; ++j)
            bf[j] = load_b_frag(BT + (size_t)(n0 + 16 * j) * ldb + k0, ldb, lane);
#pragma unroll
        for (int i = 0; i < 4; ++i)
#pragma unroll
            for (int j = 0; j < 4; ++j)
                acc[i][j] = __builtin_amdgcn_wmma_f32_16x16x32_bf16(
                    false, af[i], false, bf[j], (short)0, acc[i][j], false, false);
    }

    // Epilogue. C fragment: VGPR e, lane l -> M = e + 8*(l>>4), N = l&15.
    const int half = lane >> 4, cn = lane & 15;
#pragma unroll
    for (int i = 0; i < 4; ++i)
#pragma unroll
        for (int j = 0; j < 4; ++j) {
            int mbase = m0 + 16 * i + 8 * half;
            int n = n0 + 16 * j + cn;
            float bn = (MODE == 0 && bias) ? bias[n] : 0.0f;
#pragma unroll
            for (int e = 0; e < 8; ++e) {
                float val = acc[i][j][e];
                int m = mbase + e;
                if (MODE == 0) {
                    ((float*)C)[(size_t)m * ldc + n] = val + bn;
                } else if (MODE == 1) {
                    ((bf16_t*)C)[(size_t)m * ldc + n] = f2bf(val);
                } else if (MODE == 2) {
                    ((bf16_t*)C)[(size_t)n * ldc + m] = f2bf(val);
                } else {
                    ((bf16_t*)C)[(size_t)m * ldc + n] = f2bf(val > 0.0f ? val : 0.0f);
                }
            }
        }
}

// ---------------- row softmax: att bf16 + head-mean accumulation -------------
// One block (256 thr) per row of 4096; 16 elems per thread kept in registers.
__global__ __launch_bounds__(256)
void softmax_kernel(const float* __restrict__ scores, bf16_t* __restrict__ attb,
                    float* __restrict__ att_mean, int N, int first) {
    const int row = blockIdx.x, t = threadIdx.x;
    const float* s = scores + (size_t)row * N;
    float local[16];
    float mx = -3.4e38f;
#pragma unroll
    for (int i = 0; i < 16; ++i) {
        float v = s[t + 256 * i] * SCALE;
        local[i] = v;
        mx = fmaxf(mx, v);
    }
    __shared__ float red[256];
    red[t] = mx; __syncthreads();
    for (int off = 128; off > 0; off >>= 1) {
        if (t < off) red[t] = fmaxf(red[t], red[t + off]);
        __syncthreads();
    }
    mx = red[0]; __syncthreads();
    float sum = 0.0f;
#pragma unroll
    for (int i = 0; i < 16; ++i) { float e = __expf(local[i] - mx); local[i] = e; sum += e; }
    red[t] = sum; __syncthreads();
    for (int off = 128; off > 0; off >>= 1) {
        if (t < off) red[t] += red[t + off];
        __syncthreads();
    }
    float inv = 1.0f / red[0];
    bf16_t* arow = attb + (size_t)row * N;
    float* mrow = att_mean + (size_t)row * N;
#pragma unroll
    for (int i = 0; i < 16; ++i) {
        float p = local[i] * inv;
        arow[t + 256 * i] = f2bf(p);
        float q = 0.25f * p;                    // mean over 4 heads
        if (first) mrow[t + 256 * i] = q;
        else       mrow[t + 256 * i] += q;
    }
}

// ---------------- LayerNorm over last dim (1024), affine ---------------------
__global__ __launch_bounds__(256)
void layernorm_kernel(const float* __restrict__ h, const float* __restrict__ gamma,
                      const float* __restrict__ beta, float* __restrict__ out, int D) {
    const int row = blockIdx.x, t = threadIdx.x;
    const float* r = h + (size_t)row * D;
    float v[4]; float s = 0.0f;
#pragma unroll
    for (int i = 0; i < 4; ++i) { v[i] = r[t + 256 * i]; s += v[i]; }
    __shared__ float red[256];
    red[t] = s; __syncthreads();
    for (int off = 128; off > 0; off >>= 1) {
        if (t < off) red[t] += red[t + off];
        __syncthreads();
    }
    float mean = red[0] / (float)D; __syncthreads();
    float q = 0.0f;
#pragma unroll
    for (int i = 0; i < 4; ++i) { float d = v[i] - mean; q += d * d; }
    red[t] = q; __syncthreads();
    for (int off = 128; off > 0; off >>= 1) {
        if (t < off) red[t] += red[t + off];
        __syncthreads();
    }
    float rstd = rsqrtf(red[0] / (float)D + LN_EPS);
    float* o = out + (size_t)row * D;
#pragma unroll
    for (int i = 0; i < 4; ++i) {
        int c = t + 256 * i;
        o[c] = (v[i] - mean) * rstd * gamma[c] + beta[c];
    }
}

extern "C" void kernel_launch(void* const* d_in, const int* in_sizes, int n_in,
                              void* d_out, int out_size, void* d_ws, size_t ws_size,
                              hipStream_t stream) {
    const float* x     = (const float*)d_in[0];
    const float* Wq    = (const float*)d_in[1];
    const float* Wk    = (const float*)d_in[2];
    const float* Wv    = (const float*)d_in[3];
    const float* Wo    = (const float*)d_in[4];
    const float* bo    = (const float*)d_in[5];
    const float* gamma = (const float*)d_in[6];
    const float* beta  = (const float*)d_in[7];

    const int N = in_sizes[0] / EMB;          // 4096

    // workspace carve-up (256B aligned)
    char* ws = (char*)d_ws;
    size_t off = 0;
    auto alloc = [&](size_t bytes) { size_t o = off; off += (bytes + 255) & ~(size_t)255; return o; };
    bf16_t* xb   = (bf16_t*)(ws + alloc((size_t)N * EMB * 2));
    bf16_t* WqT  = (bf16_t*)(ws + alloc((size_t)EMB * EMB * 2));
    bf16_t* WkT  = (bf16_t*)(ws + alloc((size_t)EMB * EMB * 2));
    bf16_t* WvT  = (bf16_t*)(ws + alloc((size_t)EMB * EMB * 2));
    bf16_t* WoT  = (bf16_t*)(ws + alloc((size_t)EMB * EMB * 2));
    bf16_t* qb   = (bf16_t*)(ws + alloc((size_t)N * EMB * 2));
    bf16_t* kb   = (bf16_t*)(ws + alloc((size_t)N * EMB * 2));
    bf16_t* vT   = (bf16_t*)(ws + alloc((size_t)EMB * N * 2));   // vT[d][m]
    float*  sc   = (float*) (ws + alloc((size_t)N * N * 4));     // per-head scores
    bf16_t* attb = (bf16_t*)(ws + alloc((size_t)N * N * 2));     // per-head att bf16
    bf16_t* hb   = (bf16_t*)(ws + alloc((size_t)N * EMB * 2));   // relu(att@v)
    float*  out2 = (float*) (ws + alloc((size_t)N * EMB * 4));   // h@Wo + bo

    float* mol      = (float*)d_out;                 // [N, EMB]
    float* att_mean = (float*)d_out + (size_t)N * EMB; // [N, N]

    // 1) x -> bf16
    {
        int n = N * EMB;
        cvt_bf16_kernel<<<(n + 255) / 256, 256, 0, stream>>>(x, xb, n);
    }
    // 2) weight transposes (fp32 -> bf16 W^T)
    {
        dim3 g(EMB / 32, EMB / 32);
        transpose_cvt_kernel<<<g, 256, 0, stream>>>(Wq, WqT, EMB, EMB);
        transpose_cvt_kernel<<<g, 256, 0, stream>>>(Wk, WkT, EMB, EMB);
        transpose_cvt_kernel<<<g, 256, 0, stream>>>(Wv, WvT, EMB, EMB);
        transpose_cvt_kernel<<<g, 256, 0, stream>>>(Wo, WoT, EMB, EMB);
    }
    // 3) QKV projections (bf16 WMMA, K=1024). v written transposed.
    {
        dim3 g(EMB / 256, N / 128);
        gemm_bf16_wmma<1><<<g, 256, 0, stream>>>(xb, EMB, WqT, EMB, qb, EMB, EMB, nullptr);
        gemm_bf16_wmma<1><<<g, 256, 0, stream>>>(xb, EMB, WkT, EMB, kb, EMB, EMB, nullptr);
        gemm_bf16_wmma<2><<<g, 256, 0, stream>>>(xb, EMB, WvT, EMB, vT, N,   EMB, nullptr);
    }
    // 4) per-head attention
    for (int h = 0; h < HEADS; ++h) {
        const bf16_t* qh  = qb + h * ATT;            // lda = EMB
        const bf16_t* kh  = kb + h * ATT;            // ldb = EMB (already K-contiguous rows)
        const bf16_t* vTh = vT + (size_t)h * ATT * N; // rows d, ldb = N
        // scores = q_h @ k_h^T  (M=N=4096, K=256), fp32 out
        {
            dim3 g(N / 256, N / 128);
            gemm_bf16_wmma<0><<<g, 256, 0, stream>>>(qh, EMB, kh, EMB, sc, N, ATT, nullptr);
        }
        // softmax rows -> att bf16, accumulate head mean into d_out
        softmax_kernel<<<N, 256, 0, stream>>>(sc, attb, att_mean, N, h == 0 ? 1 : 0);
        // h_h = relu(att @ v_h)  (M=4096, N=256, K=4096), bf16 into hb columns
        {
            dim3 g(ATT / 256, N / 128);
            gemm_bf16_wmma<3><<<g, 256, 0, stream>>>(attb, N, vTh, N, hb + h * ATT, EMB, N, nullptr);
        }
    }
    // 5) output projection + bias (fp32 out)
    {
        dim3 g(EMB / 256, N / 128);
        gemm_bf16_wmma<0><<<g, 256, 0, stream>>>(hb, EMB, WoT, EMB, out2, EMB, EMB, bo);
    }
    // 6) LayerNorm -> d_out
    layernorm_kernel<<<N, 256, 0, stream>>>(out2, gamma, beta, mol, EMB);
}
